// Attention_11879879540959
// MI455X (gfx1250) — matrix operand order repaired and checked by reading, not compiled, not run on hardware
//
#include <hip/hip_runtime.h>
#include <hip/hip_bf16.h>
#include <math.h>

// Bahdanau attention, algebraically collapsed:
//   score[b,s] = enc[b,s,:] . (W2 @ V)   (+ per-b constant that cancels in softmax)
//   attn = softmax_s(score);  ctx[b] = sum_s attn[b,s] * enc[b,s,:]
// Memory-bound: 512MB of encoder_output streamed; score pass runs on WMMA.

typedef __attribute__((ext_vector_type(16))) _Float16 v16h;
typedef __attribute__((ext_vector_type(8)))  float    v8f;

#define B_SZ 2048
#define S_SZ 128
#define H_SZ 512
#define U_SZ 512

// ---------------------------------------------------------------------------
// Kernel 1: w2v[h] = sum_u W2[h,u] * V[u]   (tiny: 512x512 mat-vec)
// ---------------------------------------------------------------------------
__global__ __launch_bounds__(256) void w2v_kernel(const float* __restrict__ W2,
                                                  const float* __restrict__ V,
                                                  float* __restrict__ w2v) {
  int h = blockIdx.x * blockDim.x + threadIdx.x;
  if (h >= H_SZ) return;
  const float4* row = (const float4*)(W2 + (size_t)h * U_SZ);
  const float4* v4  = (const float4*)V;
  float acc = 0.f;
#pragma unroll 4
  for (int u = 0; u < U_SZ / 4; ++u) {
    float4 a = row[u], b = v4[u];
    acc += a.x * b.x + a.y * b.y + a.z * b.z + a.w * b.w;
  }
  w2v[h] = acc;
}

// ---------------------------------------------------------------------------
// Kernel 2: one workgroup (256 threads = 8 waves) per batch element b.
//   Stage 1: scores via v_wmma_f32_16x16x32_f16 (B-matrix = w2v replicated
//            across 16 columns; every column of C equals the score vector).
//            B fragments are pre-converted to packed f16 in LDS once.
//   Stage 2: softmax over S=128 (wave32 shuffles + LDS).
//   Stage 3: ctx[b,h] = sum_s attn[s] * enc[b,s,h] (coalesced float2 stream).
// ---------------------------------------------------------------------------
__global__ __launch_bounds__(256) void attn_fused_kernel(const float* __restrict__ enc,
                                                         const float* __restrict__ w2v_g,
                                                         float* __restrict__ ctx_out,
                                                         float* __restrict__ attn_out) {
  __shared__ __align__(32) _Float16 s_w2vh[H_SZ];  // w2v as packed f16 (1 KB)
  __shared__ float s_score[S_SZ];
  __shared__ float s_red[9];

  const int tid  = threadIdx.x;
  const int lane = tid & 31;
  const int wv   = tid >> 5;
  const int b    = blockIdx.x;
  const float* encB = enc + (size_t)b * S_SZ * H_SZ;

  // Convert w2v -> f16 once per block (2 elements per thread).
  s_w2vh[tid]       = (_Float16)w2v_g[tid];
  s_w2vh[tid + 256] = (_Float16)w2v_g[tid + 256];
  __syncthreads();

  // -------- Stage 1: WMMA score pass. Wave wv owns rows [16*wv, 16*wv+15]. --
  {
    const int m     = lane & 15;
    const int row   = wv * 16 + m;
    const float* rowp = encB + (size_t)row * H_SZ;
    // A(16x32) layout (matches B): lanes 0-15 hold K 0..15,
    // lanes 16-31 hold K 16..31 (per-lane halfs h0..h15).
    const int khalf = (lane < 16) ? 0 : 16;
    // B(32x16) layout used here (same as A): lanes 0-15 -> K 0..15, lanes 16-31 -> K 16..31.
    const int kBoff = (lane < 16) ? 0 : 16;

    v8f c = {};
#pragma unroll 2
    for (int kb = 0; kb < H_SZ; kb += 32) {
      const float4 f0 = *(const float4*)(rowp + kb + khalf);
      const float4 f1 = *(const float4*)(rowp + kb + khalf + 4);
      const float4 f2 = *(const float4*)(rowp + kb + khalf + 8);
      const float4 f3 = *(const float4*)(rowp + kb + khalf + 12);
      __builtin_prefetch(rowp + kb + 32 + khalf, 0, 1);  // global_prefetch_b8

      v16h a;
      a[0]  = (_Float16)f0.x; a[1]  = (_Float16)f0.y; a[2]  = (_Float16)f0.z; a[3]  = (_Float16)f0.w;
      a[4]  = (_Float16)f1.x; a[5]  = (_Float16)f1.y; a[6]  = (_Float16)f1.z; a[7]  = (_Float16)f1.w;
      a[8]  = (_Float16)f2.x; a[9]  = (_Float16)f2.y; a[10] = (_Float16)f2.z; a[11] = (_Float16)f2.w;
      a[12] = (_Float16)f3.x; a[13] = (_Float16)f3.y; a[14] = (_Float16)f3.z; a[15] = (_Float16)f3.w;

      // B fragment: 16 contiguous packed halfs straight from LDS, no cvt.
      const v16h bb = *(const v16h*)&s_w2vh[kb + kBoff];

      c = __builtin_amdgcn_wmma_f32_16x16x32_f16(false, a, false, bb,
                                                 (short)0, c, false, false);
    }
    // C layout: VGPR r, lanes 0-15 -> M=r; lanes 16-31 -> M=8+r (all N equal).
    if (lane == 0) {
#pragma unroll
      for (int r = 0; r < 8; ++r) s_score[wv * 16 + r] = c[r];
    }
    if (lane == 16) {
#pragma unroll
      for (int r = 0; r < 8; ++r) s_score[wv * 16 + 8 + r] = c[r];
    }
  }
  __syncthreads();

  // -------- Stage 2: softmax over s (S=128 lives in threads 0..127) --------
  float v = (tid < S_SZ) ? s_score[tid] : -3.402823466e+38f;
#pragma unroll
  for (int off = 16; off > 0; off >>= 1)
    v = fmaxf(v, __shfl_xor(v, off, 32));
  if (lane == 0) s_red[wv] = v;
  __syncthreads();
  if (tid == 0) {
    float mx = s_red[0];
    for (int i = 1; i < 8; ++i) mx = fmaxf(mx, s_red[i]);
    s_red[8] = mx;
  }
  __syncthreads();
  const float mx = s_red[8];
  float e = (tid < S_SZ) ? __expf(s_score[tid] - mx) : 0.f;
  float ps = e;
#pragma unroll
  for (int off = 16; off > 0; off >>= 1)
    ps += __shfl_xor(ps, off, 32);
  if (lane == 0) s_red[wv] = ps;
  __syncthreads();
  if (tid == 0) {
    float sm = 0.f;
    for (int i = 0; i < 8; ++i) sm += s_red[i];
    s_red[8] = 1.0f / sm;
  }
  __syncthreads();
  const float inv = s_red[8];
  if (tid < S_SZ) {
    const float w = e * inv;
    s_score[tid] = w;                           // reuse as attn weights
    attn_out[(size_t)b * S_SZ + tid] = w;       // output 2: [B,S,1]
  }
  __syncthreads();

  // -------- Stage 3: context vector (coalesced float2 per thread) ----------
  {
    const float2* base = (const float2*)encB + tid;   // columns {2*tid, 2*tid+1}
    float accx = 0.f, accy = 0.f;
#pragma unroll 8
    for (int s = 0; s < S_SZ; ++s) {
      const float w = s_score[s];
      const float2 ev = base[(size_t)s * (H_SZ / 2)];
      accx = fmaf(w, ev.x, accx);
      accy = fmaf(w, ev.y, accy);
    }
    float2 r; r.x = accx; r.y = accy;
    *(float2*)(ctx_out + (size_t)b * H_SZ + tid * 2) = r;  // output 1: [B,H]
  }
}

// ---------------------------------------------------------------------------
extern "C" void kernel_launch(void* const* d_in, const int* in_sizes, int n_in,
                              void* d_out, int out_size, void* d_ws, size_t ws_size,
                              hipStream_t stream) {
  // setup_inputs order:
  // 0: decoder_hidden_state [B,H]   (cancels in softmax -> unused)
  // 1: encoder_output       [B,S,H]
  // 2: W1 [H,U], 3: b1 [U]          (cancel -> unused)
  // 4: W2 [H,U], 5: b2 [U]          (b2 cancels)
  // 6: V  [U,1], 7: bV [1]          (bV cancels)
  const float* enc = (const float*)d_in[1];
  const float* W2  = (const float*)d_in[4];
  const float* V   = (const float*)d_in[6];

  float* out  = (float*)d_out;
  float* ctx  = out;                                // [B,H] first
  float* attn = out + (size_t)B_SZ * H_SZ;          // then [B,S,1]
  float* w2v  = (float*)d_ws;                       // 512 floats of scratch

  w2v_kernel<<<(H_SZ + 255) / 256, 256, 0, stream>>>(W2, V, w2v);
  attn_fused_kernel<<<B_SZ, 256, 0, stream>>>(enc, w2v, ctx, attn);
}
